// MultiTopologyGNN_747324310165
// MI455X (gfx1250) — matrix operand order repaired and checked by reading, not compiled, not run on hardware
//
#include <hip/hip_runtime.h>
#include <math.h>

// ---------------- CDNA5 WMMA types ----------------
typedef __attribute__((ext_vector_type(16))) __bf16 v16bf;
typedef __attribute__((ext_vector_type(8)))  __bf16 v8bf;
typedef __attribute__((ext_vector_type(8)))  float  v8f;

__device__ __forceinline__ __bf16 f2bf(float f) {
  // round-to-nearest-even f32 -> bf16 (bit-exact, no __bf16 cvt dependence)
  unsigned u = __float_as_uint(f);
  u += 0x7fffu + ((u >> 16) & 1u);
  return __builtin_bit_cast(__bf16, (unsigned short)(u >> 16));
}

__device__ __forceinline__ void atomAddF(float* p, float v) {
  (void)__hip_atomic_fetch_add(p, v, __ATOMIC_RELAXED, __HIP_MEMORY_SCOPE_AGENT);
}

__device__ __forceinline__ void atomMaxF(float* p, float v) {
  // sign-aware ordered-int trick: int max for >=0, uint min for <0
  if (v >= 0.0f) {
    (void)__hip_atomic_fetch_max((int*)p, __float_as_int(v),
                                 __ATOMIC_RELAXED, __HIP_MEMORY_SCOPE_AGENT);
  } else {
    (void)__hip_atomic_fetch_min((unsigned int*)p, (unsigned int)__float_as_int(v),
                                 __ATOMIC_RELAXED, __HIP_MEMORY_SCOPE_AGENT);
  }
}

__device__ __forceinline__ float leaky02(float x) { return x < 0.0f ? 0.2f * x : x; }

// ---------------- precision-staging kernels ----------------
__global__ void cvt_f32_bf16(const float* __restrict__ in, __bf16* __restrict__ out, int n) {
  int i = blockIdx.x * blockDim.x + threadIdx.x;
  if (i < n) out[i] = f2bf(in[i]);
}
// relu + bf16 convert (f32 copy is dead after this point in the pipeline)
__global__ void relu_cvt_bf16(const float* __restrict__ in, __bf16* __restrict__ out, int n) {
  int i = blockIdx.x * blockDim.x + threadIdx.x;
  if (i < n) out[i] = f2bf(fmaxf(in[i], 0.0f));
}
// ctx = concat(style, stress) -> bf16, n = N*512
__global__ void concat_cvt_bf16(const float* __restrict__ a, const float* __restrict__ b,
                                __bf16* __restrict__ out, int n) {
  int i = blockIdx.x * blockDim.x + threadIdx.x;
  if (i >= n) return;
  int node = i >> 9, c = i & 511;
  float v = (c < 256) ? a[(size_t)node * 256 + c] : b[(size_t)node * 256 + (c - 256)];
  out[i] = f2bf(v);
}
// W [Q x P] f32 -> Wt [P x Q] bf16 (transpose so B fragments are lane-contiguous)
__global__ void transpose_cvt_bf16(const float* __restrict__ W, __bf16* __restrict__ Wt,
                                   int Q, int P) {
  int i = blockIdx.x * blockDim.x + threadIdx.x;
  if (i >= Q * P) return;
  int q = i / P, p = i % P;
  Wt[(size_t)p * Q + q] = f2bf(W[i]);
}

// ---------------- GEMM: C = act(A @ W [+ bias]) via v_wmma_f32_16x16x32_bf16 ----
// A: [M x Q] bf16 row-major. Wt: [P x Q] bf16 (W transposed). One wave computes a
// 16x64 tile (1 A fragment reused by 4 WMMAs). All 10 b128 loads of an iteration
// are issued before the first WMMA so the backend can stagger s_wait_loadcnt and
// overlap WMMA with the remaining loads.
template<int Q>
__global__ void gemm_wmma_bf16(const __bf16* __restrict__ A,
                               const __bf16* __restrict__ Wt,
                               const float* __restrict__ bias,
                               float* __restrict__ C,
                               int M, int P, int act)
{
  const int lane = threadIdx.x & 31;
  const int wave = blockIdx.x * (blockDim.x >> 5) + (threadIdx.x >> 5);
  const int tilesN = P >> 6;                    // 64-wide wave tiles
  const int totalTiles = (M >> 4) * tilesN;
  const int tile = (wave < totalTiles) ? wave : 0;  // clamped dups rewrite tile 0 (identical data)
  const int tm  = tile / tilesN;
  const int cn0 = (tile % tilesN) << 6;

  const int hiHalf = lane >> 4;                 // 0 or 1
  const int kA0 = hiHalf << 4;                  // A: lanes 16-31 hold K+16..K+31
  const int kB0 = hiHalf << 4;                  // B: lanes 16-31 hold K+16..K+31

  const __bf16* aRow  = A + (size_t)((tm << 4) + (lane & 15)) * Q;
  const __bf16* bCol0 = Wt + (size_t)(cn0 + (lane & 15)) * Q + kB0;

  v8f acc[4] = {{}, {}, {}, {}};
#pragma unroll 2
  for (int kk = 0; kk < Q; kk += 32) {
    // ---- issue ALL loads for this k-chunk first (2 A + 8 B b128 loads) ----
    v8bf alo = *(const v8bf*)(aRow + kk + kA0);
    v8bf ahi = *(const v8bf*)(aRow + kk + 8 + kA0);
    v8bf bl[4], bh[4];
#pragma unroll
    for (int j = 0; j < 4; ++j) {
      const __bf16* bp = bCol0 + (size_t)(j * 16) * Q + kk;
      bl[j] = *(const v8bf*)(bp);
      bh[j] = *(const v8bf*)(bp + 8);
    }
    v16bf a = __builtin_shufflevector(alo, ahi, 0, 1, 2, 3, 4, 5, 6, 7,
                                                8, 9, 10, 11, 12, 13, 14, 15);
    // ---- 4 WMMAs, waits stagger as loads retire in order ----
#pragma unroll
    for (int j = 0; j < 4; ++j) {
      v16bf b = __builtin_shufflevector(bl[j], bh[j], 0, 1, 2, 3, 4, 5, 6, 7,
                                                      8, 9, 10, 11, 12, 13, 14, 15);
      acc[j] = __builtin_amdgcn_wmma_f32_16x16x32_bf16(false, a, false, b,
                                                       (short)0, acc[j], false, false);
    }
  }

  // C 16x16 f32 layout: VGPR r -> M = r + 8*hiHalf, N = lane&15
  const int rowC0 = (tm << 4) + (hiHalf << 3);
#pragma unroll
  for (int j = 0; j < 4; ++j) {
    const int colC = cn0 + j * 16 + (lane & 15);
    const float bs = bias ? bias[colC] : 0.0f;
#pragma unroll
    for (int r = 0; r < 8; ++r) {
      float v = acc[j][r] + bs;
      if (act == 2) v = tanhf(v);
      C[(size_t)(rowC0 + r) * P + colC] = v;
    }
  }
}

// ---------------- degree / normalization ----------------
__global__ void deg_init(float* deg, int n) {
  int i = blockIdx.x * blockDim.x + threadIdx.x;
  if (i < n) deg[i] = 1.0f;   // self-loop
}
__global__ void deg_edges(const int* __restrict__ dst, float* deg, int e) {
  int i = blockIdx.x * blockDim.x + threadIdx.x;
  if (i < e) atomAddF(&deg[dst[i]], 1.0f);
}
__global__ void deg_to_dinv(float* deg, int n) {
  int i = blockIdx.x * blockDim.x + threadIdx.x;
  if (i < n) deg[i] = rsqrtf(deg[i]);   // deg >= 1 always (self-loop)
}

// ---------------- GCN aggregation ----------------
__global__ void gcn_self_bias(const float* __restrict__ h, const float* __restrict__ dinv,
                              const float* __restrict__ b, float* __restrict__ out, int n) {
  int idx = blockIdx.x * blockDim.x + threadIdx.x;
  if (idx >= n) return;
  int i = idx >> 8, c = idx & 255;
  float di = dinv[i];
  out[idx] = b[c] + di * di * h[idx];
}

__global__ void gcn_scatter(const float* __restrict__ h, const int* __restrict__ src,
                            const int* __restrict__ dst, const float* __restrict__ dinv,
                            float* __restrict__ out, int E_) {
  int t = blockIdx.x * blockDim.x + threadIdx.x;
  int e = t >> 5, lane = t & 31;
  if (e >= E_) return;
  int s = src[e], d = dst[e];
  float wgt = dinv[s] * dinv[d];
  const float* hs = h + (size_t)s * 256 + lane * 8;
  float* od = out + (size_t)d * 256 + lane * 8;
  float4 v0 = *(const float4*)hs;
  float4 v1 = *(const float4*)(hs + 4);
  atomAddF(od + 0, v0.x * wgt); atomAddF(od + 1, v0.y * wgt);
  atomAddF(od + 2, v0.z * wgt); atomAddF(od + 3, v0.w * wgt);
  atomAddF(od + 4, v1.x * wgt); atomAddF(od + 5, v1.y * wgt);
  atomAddF(od + 6, v1.z * wgt); atomAddF(od + 7, v1.w * wgt);
}

// ---------------- GAT ----------------
__global__ void gat_logits(const float* __restrict__ h, const float* __restrict__ asrc,
                           const float* __restrict__ adst, float* __restrict__ ls,
                           float* __restrict__ ld, float* __restrict__ m, int N_) {
  int t = blockIdx.x * blockDim.x + threadIdx.x;
  if (t >= N_ * 8) return;
  int i = t >> 3, hd = t & 7;
  const float* hp = h + (size_t)i * 256 + hd * 32;
  const float* as = asrc + hd * 32;
  const float* ad = adst + hd * 32;
  float s1 = 0.0f, s2 = 0.0f;
#pragma unroll 8
  for (int c = 0; c < 32; ++c) { float v = hp[c]; s1 += v * as[c]; s2 += v * ad[c]; }
  ls[t] = s1; ld[t] = s2;
  m[t] = leaky02(s1 + s2);   // self-loop initializes segment max
}

__global__ void gat_edge_max(const int* __restrict__ src, const int* __restrict__ dst,
                             const float* __restrict__ ls, const float* __restrict__ ld,
                             float* __restrict__ m, int E_) {
  int t = blockIdx.x * blockDim.x + threadIdx.x;
  if (t >= E_ * 8) return;
  int e = t >> 3, hd = t & 7;
  int s = src[e], d = dst[e];
  atomMaxF(&m[d * 8 + hd], leaky02(ls[s * 8 + hd] + ld[d * 8 + hd]));
}

__global__ void gat_denom_init(const float* __restrict__ ls, const float* __restrict__ ld,
                               const float* __restrict__ m, float* __restrict__ dn, int n8) {
  int t = blockIdx.x * blockDim.x + threadIdx.x;
  if (t >= n8) return;
  dn[t] = __expf(leaky02(ls[t] + ld[t]) - m[t]);   // self-loop term
}

__global__ void gat_edge_denom(const int* __restrict__ src, const int* __restrict__ dst,
                               const float* __restrict__ ls, const float* __restrict__ ld,
                               const float* __restrict__ m, float* __restrict__ dn, int E_) {
  int t = blockIdx.x * blockDim.x + threadIdx.x;
  if (t >= E_ * 8) return;
  int e = t >> 3, hd = t & 7;
  int s = src[e], d = dst[e];
  float v = leaky02(ls[s * 8 + hd] + ld[d * 8 + hd]);
  atomAddF(&dn[d * 8 + hd], __expf(v - m[d * 8 + hd]));
}

__global__ void gat_out_init(const float* __restrict__ h, const float* __restrict__ ls,
                             const float* __restrict__ ld, const float* __restrict__ m,
                             const float* __restrict__ dn, const float* __restrict__ b,
                             float* __restrict__ out, int n) {
  int idx = blockIdx.x * blockDim.x + threadIdx.x;
  if (idx >= n) return;
  int i = idx >> 8, c = idx & 255, hd = c >> 5;
  float e = leaky02(ls[i * 8 + hd] + ld[i * 8 + hd]);
  float alpha = __expf(e - m[i * 8 + hd]) / dn[i * 8 + hd];
  out[idx] = b[c] + h[idx] * alpha;
}

__global__ void gat_edge_scatter(const float* __restrict__ h, const int* __restrict__ src,
                                 const int* __restrict__ dst, const float* __restrict__ ls,
                                 const float* __restrict__ ld, const float* __restrict__ m,
                                 const float* __restrict__ dn, float* __restrict__ out, int E_) {
  int t = blockIdx.x * blockDim.x + threadIdx.x;
  int e = t >> 5, lane = t & 31;
  if (e >= E_) return;
  int s = src[e], d = dst[e];
  int hd = lane >> 2;   // 8 cols per lane, all inside one 32-col head
  float v = leaky02(ls[s * 8 + hd] + ld[d * 8 + hd]);
  float alpha = __expf(v - m[d * 8 + hd]) / dn[d * 8 + hd];
  const float* hs = h + (size_t)s * 256 + lane * 8;
  float* od = out + (size_t)d * 256 + lane * 8;
  float4 v0 = *(const float4*)hs;
  float4 v1 = *(const float4*)(hs + 4);
  atomAddF(od + 0, v0.x * alpha); atomAddF(od + 1, v0.y * alpha);
  atomAddF(od + 2, v0.z * alpha); atomAddF(od + 3, v0.w * alpha);
  atomAddF(od + 4, v1.x * alpha); atomAddF(od + 5, v1.y * alpha);
  atomAddF(od + 6, v1.z * alpha); atomAddF(od + 7, v1.w * alpha);
}

// ---------------- LayerNorm + weighted combine (fused epilogue) ----------------
// one wave32 per node; out (+)= w[node,k] * LN(h[node])
__global__ void ln_accum(const float* __restrict__ hin, const float* __restrict__ g,
                         const float* __restrict__ bta, const float* __restrict__ w,
                         int kIdx, int accum, float* __restrict__ out, int N_) {
  int node = blockIdx.x * (blockDim.x >> 5) + (threadIdx.x >> 5);
  int lane = threadIdx.x & 31;
  if (node >= N_) return;
  const float* hp = hin + (size_t)node * 256;
  float v[8], s = 0.0f, ss = 0.0f;
#pragma unroll
  for (int j = 0; j < 8; ++j) {
    v[j] = hp[lane * 8 + j];
    s += v[j]; ss += v[j] * v[j];
  }
#pragma unroll
  for (int off = 16; off; off >>= 1) {
    s  += __shfl_xor(s,  off, 32);
    ss += __shfl_xor(ss, off, 32);
  }
  float mu  = s * (1.0f / 256.0f);
  float var = ss * (1.0f / 256.0f) - mu * mu;
  float inv = rsqrtf(var + 1e-5f);
  float wk = w[node * 4 + kIdx];
  float* op = out + (size_t)node * 256 + lane * 8;
#pragma unroll
  for (int j = 0; j < 8; ++j) {
    int c = lane * 8 + j;
    float r = wk * ((v[j] - mu) * inv * g[c] + bta[c]);
    op[j] = accum ? (op[j] + r) : r;
  }
}

// ---------------- fusion logits + softmax over K=4 ----------------
__global__ void fus_weights(const float* __restrict__ t, const float* __restrict__ W2,
                            const float* __restrict__ b2, float* __restrict__ w, int N_) {
  int i = blockIdx.x * blockDim.x + threadIdx.x;
  if (i >= N_) return;
  const float* tp = t + (size_t)i * 256;
  float l0 = b2[0], l1 = b2[1], l2 = b2[2], l3 = b2[3];
  for (int c = 0; c < 256; ++c) {
    float v = tp[c];
    const float* r = W2 + c * 4;
    l0 += v * r[0]; l1 += v * r[1]; l2 += v * r[2]; l3 += v * r[3];
  }
  float mx = fmaxf(fmaxf(l0, l1), fmaxf(l2, l3));
  float e0 = __expf(l0 - mx), e1 = __expf(l1 - mx), e2 = __expf(l2 - mx), e3 = __expf(l3 - mx);
  float inv = 1.0f / (e0 + e1 + e2 + e3);
  w[i * 4 + 0] = e0 * inv; w[i * 4 + 1] = e1 * inv;
  w[i * 4 + 2] = e2 * inv; w[i * 4 + 3] = e3 * inv;
}

// ---------------- host ----------------
extern "C" void kernel_launch(void* const* d_in, const int* in_sizes, int n_in,
                              void* d_out, int out_size, void* d_ws, size_t ws_size,
                              hipStream_t stream) {
  (void)n_in; (void)out_size; (void)ws_size;
  const int N = in_sizes[0] / 512;   // 50000
  const int E = in_sizes[3] / 2;     // 600000
  const int HID = 256;

  const float* x      = (const float*)d_in[0];
  const float* style  = (const float*)d_in[1];
  const float* stress = (const float*)d_in[2];
  const int* eidx[4]  = {(const int*)d_in[3], (const int*)d_in[4],
                         (const int*)d_in[5], (const int*)d_in[6]};
  const float* gW0 = (const float*)d_in[7];
  const float* gb0 = (const float*)d_in[8];
  const float* gW1 = (const float*)d_in[9];
  const float* gb1 = (const float*)d_in[10];
  const float* gW2 = (const float*)d_in[11];
  const float* gb2 = (const float*)d_in[12];
  const float* aW  = (const float*)d_in[13];
  const float* aas = (const float*)d_in[14];
  const float* aad = (const float*)d_in[15];
  const float* ab  = (const float*)d_in[16];
  const float* lng = (const float*)d_in[17];
  const float* lnb = (const float*)d_in[18];
  const float* fW1 = (const float*)d_in[19];
  const float* fb1 = (const float*)d_in[20];
  const float* fW2 = (const float*)d_in[21];
  const float* fb2 = (const float*)d_in[22];
  float* out = (float*)d_out;

  // workspace carve (all buffers fully written before read)
  char* wsp = (char*)d_ws;
  float*  t0   = (float*)wsp;  wsp += (size_t)N * HID * 4;   // GEMM out (f32)
  float*  t1   = (float*)wsp;  wsp += (size_t)N * HID * 4;   // aggregation out (f32)
  float*  wgt  = (float*)wsp;  wsp += (size_t)N * 4 * 4;     // fusion weights
  float*  deg  = (float*)wsp;  wsp += (size_t)N * 4;         // degree -> dinv in place
  float*  ls   = (float*)wsp;  wsp += (size_t)N * 8 * 4;
  float*  ld   = (float*)wsp;  wsp += (size_t)N * 8 * 4;
  float*  mm   = (float*)wsp;  wsp += (size_t)N * 8 * 4;
  float*  dn   = (float*)wsp;  wsp += (size_t)N * 8 * 4;
  __bf16* xbf  = (__bf16*)wsp; wsp += (size_t)N * 512 * 2;   // x / ctx in bf16
  __bf16* abf  = (__bf16*)wsp; wsp += (size_t)N * HID * 2;   // activations in bf16
  __bf16* wtbf = (__bf16*)wsp; wsp += (size_t)512 * HID * 2; // transposed weight (bf16)

  const dim3 blk(256);
  auto g1 = [](long n) { return dim3((unsigned)((n + 255) / 256)); };
  // 16x64 tile per wave, 8 waves per block
  const dim3 gGemm(((N / 16) * (HID / 64) + 7) / 8);

  // ---- 1) fusion weights first (independent of branch outputs) ----
  concat_cvt_bf16<<<g1((long)N * 512), blk, 0, stream>>>(style, stress, xbf, N * 512);
  transpose_cvt_bf16<<<g1(512 * HID), blk, 0, stream>>>(fW1, wtbf, 512, HID);
  gemm_wmma_bf16<512><<<gGemm, blk, 0, stream>>>(xbf, wtbf, fb1, t0, N, HID, /*act=*/2);
  fus_weights<<<g1(N), blk, 0, stream>>>(t0, fW2, fb2, wgt, N);

  // x -> bf16 once, reused by all 4 branches' first layer
  cvt_f32_bf16<<<g1((long)N * 512), blk, 0, stream>>>(x, xbf, N * 512);

  // ---- 2) branches, combining into out via ln_accum epilogue ----
  for (int k = 0; k < 4; ++k) {
    const int* src = eidx[k];
    const int* dst = eidx[k] + E;

    deg_init<<<g1(N), blk, 0, stream>>>(deg, N);
    deg_edges<<<g1(E), blk, 0, stream>>>(dst, deg, E);
    deg_to_dinv<<<g1(N), blk, 0, stream>>>(deg, N);

    // GCN layer 0: xbf(512) @ gW0 -> t0 -> aggregate -> t1 -> relu+cvt -> abf
    transpose_cvt_bf16<<<g1(512 * HID), blk, 0, stream>>>(gW0 + (size_t)k * 512 * HID,
                                                          wtbf, 512, HID);
    gemm_wmma_bf16<512><<<gGemm, blk, 0, stream>>>(xbf, wtbf, nullptr, t0, N, HID, 0);
    gcn_self_bias<<<g1((long)N * HID), blk, 0, stream>>>(t0, deg, gb0 + k * HID, t1, N * HID);
    gcn_scatter<<<g1((long)E * 32), blk, 0, stream>>>(t0, src, dst, deg, t1, E);
    relu_cvt_bf16<<<g1((long)N * HID), blk, 0, stream>>>(t1, abf, N * HID);

    // GCN layers 1,2: abf(256) @ W -> t0 -> aggregate -> t1 -> relu+cvt -> abf
    const float* Ws[2] = {gW1 + (size_t)k * HID * HID, gW2 + (size_t)k * HID * HID};
    const float* bs[2] = {gb1 + k * HID, gb2 + k * HID};
    for (int l = 0; l < 2; ++l) {
      transpose_cvt_bf16<<<g1(HID * HID), blk, 0, stream>>>(Ws[l], wtbf, HID, HID);
      gemm_wmma_bf16<256><<<gGemm, blk, 0, stream>>>(abf, wtbf, nullptr, t0, N, HID, 0);
      gcn_self_bias<<<g1((long)N * HID), blk, 0, stream>>>(t0, deg, bs[l], t1, N * HID);
      gcn_scatter<<<g1((long)E * 32), blk, 0, stream>>>(t0, src, dst, deg, t1, E);
      relu_cvt_bf16<<<g1((long)N * HID), blk, 0, stream>>>(t1, abf, N * HID);
    }

    // GAT: h = abf @ aW -> t0 (f32) ; attention ; output -> t1
    transpose_cvt_bf16<<<g1(HID * HID), blk, 0, stream>>>(aW + (size_t)k * HID * HID,
                                                          wtbf, HID, HID);
    gemm_wmma_bf16<256><<<gGemm, blk, 0, stream>>>(abf, wtbf, nullptr, t0, N, HID, 0);
    gat_logits<<<g1((long)N * 8), blk, 0, stream>>>(t0, aas + k * HID, aad + k * HID,
                                                    ls, ld, mm, N);
    gat_edge_max<<<g1((long)E * 8), blk, 0, stream>>>(src, dst, ls, ld, mm, E);
    gat_denom_init<<<g1((long)N * 8), blk, 0, stream>>>(ls, ld, mm, dn, N * 8);
    gat_edge_denom<<<g1((long)E * 8), blk, 0, stream>>>(src, dst, ls, ld, mm, dn, E);
    gat_out_init<<<g1((long)N * HID), blk, 0, stream>>>(t0, ls, ld, mm, dn,
                                                        ab + k * HID, t1, N * HID);
    gat_edge_scatter<<<g1((long)E * 32), blk, 0, stream>>>(t0, src, dst, ls, ld, mm, dn, t1, E);

    // LayerNorm + weighted accumulate into d_out (k==0 writes, k>0 adds)
    ln_accum<<<dim3((N + 7) / 8), blk, 0, stream>>>(t1, lng + k * HID, lnb + k * HID,
                                                    wgt, k, k > 0 ? 1 : 0, out, N);
  }
}
